// ExcitedMACE_80290118631832
// MI455X (gfx1250) — matrix operand order repaired and checked by reading, not compiled, not run on hardware
//
#include <hip/hip_runtime.h>
#include <hip/hip_bf16.h>
#include <math.h>

typedef __attribute__((ext_vector_type(16))) _Float16 v16h;
typedef __attribute__((ext_vector_type(2)))  _Float16 v2h;
typedef __attribute__((ext_vector_type(8)))  float    v8f;

// ---------------- problem constants ----------------
constexpr int   NN   = 10000;
constexpr int   EE   = 160000;
constexpr int   CC   = 64;
constexpr int   NZC  = 10;
constexpr int   GG   = 32;
constexpr float R_MAX   = 5.0f;
constexpr float AVG_NEI = 16.0f;
constexpr int   NTILE_E = EE / 16;   // 10000
constexpr int   NTILE_N = NN / 16;   // 625

// d_out layout (float32): e_tot[32*3] | node_e[10000*3] | nacs[10000*9] | dip[10000*18]
constexpr int OUT_NODEE = 96;
constexpr int OUT_NACS  = 30096;
constexpr int OUT_DIP   = 120096;
constexpr int OUT_TOT   = 300096;

// workspace byte offsets
constexpr size_t HOFF = 0;            // h      : N*16*64 f32 = 40,960,000 B
constexpr size_t AOFF = 40960000;     // A raw  : 40,960,000 B
constexpr size_t MOFF = 81920000;     // A mixed: 40,960,000 B
constexpr size_t YOFF = 122880000;    // Y      : E*16 f32 = 10,240,000 B
constexpr size_t FOFF = 133120000;    // feats  : E*8  f32 =  5,120,000 B
constexpr size_t WOFF = 138240000;    // f16 weights: 77,824 halves
// f16 weight element offsets (in halves) inside WOFF region
//   W1t[2][64][32] @0 | W2t[2][64][64] @4096 | W3t[2][256][64] @12288 | Wmixt[2][4][64][64] @45056

#define LDS_FENCE() asm volatile("s_wait_dscnt 0" ::: "memory")

// native v_rcp_f32 based silu: 1 trans op instead of an IEEE div sequence
__device__ __forceinline__ float siluf(float x) {
    return x * __builtin_amdgcn_rcpf(1.0f + __expf(-x));
}

// B-fragment (f16 32x16) from transposed [N][K] storage (shared or global).
// layout: lane n = lane&15, K = (v>>2)*16 + hi*8 + (v&3)*2 + j  (j packed in consecutive halves)
__device__ __forceinline__ v16h load_bfrag(const _Float16* W, int n, int kstride, int kofs, int hi) {
    v16h b;
    const v2h* p = (const v2h*)(W + (size_t)n * kstride + kofs + (hi << 3));
#pragma unroll
    for (int v = 0; v < 8; ++v) { v2h q = p[v + ((v >> 2) << 2)]; b[2 * v] = q[0]; b[2 * v + 1] = q[1]; }
    return b;
}

// A-fragment (f16 16x32) from a contiguous 64-float row (K-major), K offset kofs.
// layout: lane m = lane&15 picks the row (caller passes row ptr), K = (v>>2)*16 + hi*8 + (v&3)*2 + j
__device__ __forceinline__ v16h build_afrag(const float* row, int hi, int kofs) {
    v16h a;
#pragma unroll
    for (int v = 0; v < 8; ++v)
#pragma unroll
        for (int j = 0; j < 2; ++j) {
            int K = kofs + ((v >> 2) << 4) + (hi << 3) + ((v & 3) << 1) + j;
            a[2 * v + j] = (_Float16)row[K];
        }
    return a;
}

// ---------------- utility kernels ----------------
__global__ void k_zero(float* __restrict__ p, int n) {
    for (int i = blockIdx.x * blockDim.x + threadIdx.x; i < n; i += gridDim.x * blockDim.x)
        p[i] = 0.0f;
}

__global__ void k_convert_weights(const float* __restrict__ W_r1, const float* __restrict__ W_r2,
                                  const float* __restrict__ W_r3, const float* __restrict__ W_mix,
                                  _Float16* __restrict__ wh) {
    int idx = blockIdx.x * blockDim.x + threadIdx.x;
    if (idx >= 77824) return;
    float v;
    if (idx < 4096) {                       // W1t[i][n][k]: k<8 from W_r1[i][k][n], else 0
        int i = idx >> 11, r = idx & 2047, n = r >> 5, k = r & 31;
        v = (k < 8) ? W_r1[i * 512 + k * 64 + n] : 0.0f;
    } else if (idx < 12288) {               // W2t[i][n][k] = W_r2[i][k][n]
        int t = idx - 4096, i = t >> 12, r = t & 4095, n = r >> 6, k = r & 63;
        v = W_r2[i * 4096 + k * 64 + n];
    } else if (idx < 45056) {               // W3t[i][n][k] = W_r3[i][k][n], n<256
        int t = idx - 12288, i = t >> 14, r = t & 16383, n = r >> 6, k = r & 63;
        v = W_r3[i * 16384 + k * 256 + n];
    } else {                                // Wmixt[i][l][d][c] = W_mix[i][l][c][d]
        int t = idx - 45056, i = t >> 14, r = t & 16383, l = r >> 12;
        int rr = r & 4095, d = rr >> 6, c = rr & 63;
        v = W_mix[((i * 4 + l) * 64 + c) * 64 + d];
    }
    wh[idx] = (_Float16)v;
}

__global__ void k_init_h(float* __restrict__ h, const float* __restrict__ attrs,
                         const float* __restrict__ W_embed) {
    int idx = blockIdx.x * blockDim.x + threadIdx.x;
    if (idx >= NN * 1024) return;
    int c = idx & 63, lm = (idx >> 6) & 15, n = idx >> 10;
    float v = 0.0f;
    if (lm == 0) {
#pragma unroll
        for (int z = 0; z < NZC; ++z) v += attrs[n * NZC + z] * W_embed[z * 64 + c];
    }
    h[idx] = v;
}

__global__ void k_init_nodes(const float* __restrict__ attrs, const float* __restrict__ ae,
                             const int* __restrict__ batch, float* __restrict__ out) {
    int n = blockIdx.x * blockDim.x + threadIdx.x;
    if (n >= NN) return;
    float e0 = 0.0f;
#pragma unroll
    for (int z = 0; z < NZC; ++z) e0 += attrs[n * NZC + z] * ae[z];
    int b = batch[n];
#pragma unroll
    for (int j = 0; j < 3; ++j) {
        out[OUT_NODEE + n * 3 + j] = e0;
        atomicAdd(&out[b * 3 + j], e0);
    }
}

// ---------------- edge geometry: spherical harmonics + bessel*envelope ----------------
__global__ void k_edge_geom(const float* __restrict__ P, const float* __restrict__ SH,
                            const int* __restrict__ EI, float* __restrict__ Yw,
                            float* __restrict__ Fw) {
    int e = blockIdx.x * blockDim.x + threadIdx.x;
    if (e >= EE) return;
    int s = EI[e], r = EI[EE + e];
    float vx = P[r * 3 + 0] - P[s * 3 + 0] + SH[e * 3 + 0];
    float vy = P[r * 3 + 1] - P[s * 3 + 1] + SH[e * 3 + 1];
    float vz = P[r * 3 + 2] - P[s * 3 + 2] + SH[e * 3 + 2];
    float d2  = vx * vx + vy * vy + vz * vz + 1e-12f;
    float inv = __builtin_amdgcn_rsqf(d2);   // native v_rsq_f32: 1/len
    float len = d2 * inv;                    // len = d2 * (1/sqrt(d2)) = sqrt(d2)
    float x = vx * inv, y = vy * inv, z = vz * inv;
    float x2 = x * x, y2 = y * y, z2 = z * z;
    const float s3 = 1.7320508f, s15 = 3.8729833f, s5 = 2.2360680f, s7 = 2.6457513f;
    const float s105 = 10.2469508f, s35_8 = 2.0916500f, s21_8 = 1.6201851f;
    float* Y = Yw + (size_t)e * 16;
    Y[0]  = 1.0f;
    Y[1]  = s3 * x;  Y[2] = s3 * y;  Y[3] = s3 * z;
    Y[4]  = s15 * x * y;  Y[5] = s15 * y * z;
    Y[6]  = 0.5f * s5 * (3.0f * z2 - 1.0f);
    Y[7]  = s15 * x * z;  Y[8] = 0.5f * s15 * (x2 - y2);
    Y[9]  = s35_8 * y * (3.0f * x2 - y2);
    Y[10] = s105 * x * y * z;
    Y[11] = s21_8 * y * (5.0f * z2 - 1.0f);
    Y[12] = 0.5f * s7 * z * (5.0f * z2 - 3.0f);
    Y[13] = s21_8 * x * (5.0f * z2 - 1.0f);
    Y[14] = 0.5f * s105 * z * (x2 - y2);
    Y[15] = s35_8 * x * (x2 - 3.0f * y2);

    float xr = len * (1.0f / R_MAX);
    float xr5 = xr * xr * xr * xr * xr;
    float env = (xr < 1.0f) ? (1.0f - 21.0f * xr5 + 35.0f * xr5 * xr - 15.0f * xr5 * xr * xr) : 0.0f;
    float pref = 0.63245553f * inv * env;  // sqrt(2/R_MAX)=sqrt(0.4)
    const float PI = 3.14159265358979f;
#pragma unroll
    for (int k = 1; k <= 8; ++k) Fw[(size_t)e * 8 + (k - 1)] = pref * __sinf((float)k * PI * xr);
}

// ---------------- fused WMMA edge MLP + message + atomic scatter ----------------
__global__ __launch_bounds__(64) void k_edge_mlp(const float* __restrict__ Fw,
                                                 const float* __restrict__ Yw,
                                                 const float* __restrict__ HB,
                                                 const int* __restrict__ EI,
                                                 float* __restrict__ AB,
                                                 const _Float16* __restrict__ WH, int iter) {
    __shared__ _Float16 sW1[64 * 32];
    __shared__ _Float16 sW2[64 * 64];
    __shared__ _Float16 sW3[256 * 64];
    __shared__ float    sPer[2 * 2464];  // per wave: feat128 | Y256 | h0 1024 | act 1024 | idx 32

    const _Float16* gW1 = WH + iter * 2048;
    const _Float16* gW2 = WH + 4096 + iter * 4096;
    const _Float16* gW3 = WH + 12288 + iter * 16384;
    for (int t = threadIdx.x; t < 1024; t += 64) ((unsigned*)sW1)[t] = ((const unsigned*)gW1)[t];
    for (int t = threadIdx.x; t < 2048; t += 64) ((unsigned*)sW2)[t] = ((const unsigned*)gW2)[t];
    for (int t = threadIdx.x; t < 8192; t += 64) ((unsigned*)sW3)[t] = ((const unsigned*)gW3)[t];
    __syncthreads();

    const int lane = threadIdx.x & 31;
    const int wv   = threadIdx.x >> 5;
    const int m    = lane & 15;
    const int hi   = lane >> 4;

    float* featb = sPer + wv * 2464;
    float* Yb    = featb + 128;
    float* h0b   = Yb + 256;
    float* actb  = h0b + 1024;
    int*   idxb  = (int*)(actb + 1024);

    const int stride = gridDim.x * 2;
    for (int tile = blockIdx.x * 2 + wv; tile < NTILE_E; tile += stride) {
        const int eb = tile << 4;
        if (tile + stride < NTILE_E)
            __builtin_prefetch(Fw + (size_t)(tile + stride) * 128, 0, 0);

        // stage feats, Y, indices
#pragma unroll
        for (int t = 0; t < 4; ++t) { int id = t * 32 + lane; featb[id] = Fw[(size_t)eb * 8 + id]; }
#pragma unroll
        for (int t = 0; t < 8; ++t) { int id = t * 32 + lane; Yb[id] = Yw[(size_t)eb * 16 + id]; }
        idxb[lane] = (lane < 16) ? EI[eb + lane] : EI[EE + eb + (lane - 16)];
        LDS_FENCE();
        // gather h[send,0,:]
#pragma unroll
        for (int t = 0; t < 32; ++t) {
            int id = t * 32 + lane;
            h0b[id] = HB[(size_t)idxb[id >> 6] * 1024 + (id & 63)];
        }
        LDS_FENCE();

        // ---- layer 1: (16x8) @ (8x64), K padded to 32 ----
        v16h a0;
#pragma unroll
        for (int v = 0; v < 8; ++v)
#pragma unroll
            for (int j = 0; j < 2; ++j) {
                int K = ((v >> 2) << 4) + (hi << 3) + ((v & 3) << 1) + j;
                a0[2 * v + j] = (K < 8) ? (_Float16)featb[m * 8 + K] : (_Float16)0.0f;
            }
#pragma unroll
        for (int nt = 0; nt < 4; ++nt) {
            v8f acc = {};
            v16h b = load_bfrag(sW1, nt * 16 + m, 32, 0, hi);
            acc = __builtin_amdgcn_wmma_f32_16x16x32_f16(false, a0, false, b, (short)0, acc, false, false);
#pragma unroll
            for (int r = 0; r < 8; ++r)
                actb[(r + (hi << 3)) * 64 + nt * 16 + m] = siluf(acc[r]);
        }
        LDS_FENCE();

        // ---- layer 2: (16x64) @ (64x64) ----
        v16h a1lo = build_afrag(actb + m * 64, hi, 0);
        v16h a1hi = build_afrag(actb + m * 64, hi, 32);
#pragma unroll
        for (int nt = 0; nt < 4; ++nt) {
            v8f acc = {};
            v16h b0 = load_bfrag(sW2, nt * 16 + m, 64, 0, hi);
            acc = __builtin_amdgcn_wmma_f32_16x16x32_f16(false, a1lo, false, b0, (short)0, acc, false, false);
            v16h b1 = load_bfrag(sW2, nt * 16 + m, 64, 32, hi);
            acc = __builtin_amdgcn_wmma_f32_16x16x32_f16(false, a1hi, false, b1, (short)0, acc, false, false);
#pragma unroll
            for (int r = 0; r < 8; ++r)
                actb[(r + (hi << 3)) * 64 + nt * 16 + m] = siluf(acc[r]);
        }
        LDS_FENCE();

        // ---- layer 3: (16x64) @ (64x256) fused with msg + scatter ----
        v16h a2lo = build_afrag(actb + m * 64, hi, 0);
        v16h a2hi = build_afrag(actb + m * 64, hi, 32);
#pragma unroll
        for (int nt = 0; nt < 16; ++nt) {
            const int l = nt >> 2;          // L index of this 16-col tile
            v8f acc = {};
            v16h b0 = load_bfrag(sW3, nt * 16 + m, 64, 0, hi);
            acc = __builtin_amdgcn_wmma_f32_16x16x32_f16(false, a2lo, false, b0, (short)0, acc, false, false);
            v16h b1 = load_bfrag(sW3, nt * 16 + m, 64, 32, hi);
            acc = __builtin_amdgcn_wmma_f32_16x16x32_f16(false, a2hi, false, b1, (short)0, acc, false, false);
            const int c  = ((nt & 3) << 4) + m;   // channel 0..63
            const int ls = l * l;                 // first lm with this l
            const int lc = 2 * l + 1;             // number of lm's
#pragma unroll
            for (int r = 0; r < 8; ++r) {
                int mm = r + (hi << 3);           // edge-in-tile
                float v0 = acc[r] * h0b[mm * 64 + c] * (1.0f / AVG_NEI);
                float* dst = AB + (size_t)idxb[16 + mm] * 1024 + c;
                for (int q = 0; q < lc; ++q)
                    atomicAdd(dst + (ls + q) * 64, v0 * Yb[mm * 16 + ls + q]);
            }
        }
    }
}

// ---------------- WMMA channel mixing: Amix[n,lm,:] = A[n,lm,:] @ Wmix[l] ----------------
__global__ __launch_bounds__(128) void k_mix(const float* __restrict__ Ab, float* __restrict__ Am,
                                             const _Float16* __restrict__ WH, int iter) {
    const int lane  = threadIdx.x & 31;
    const int waveG = (blockIdx.x * blockDim.x + threadIdx.x) >> 5;
    if (waveG >= NTILE_N * 16) return;
    const int tile = waveG >> 4;
    const int lm   = waveG & 15;
    const int l    = (lm == 0) ? 0 : (lm < 4) ? 1 : (lm < 9) ? 2 : 3;
    const int m    = lane & 15;
    const int hi   = lane >> 4;

    const float* src = Ab + (size_t)(tile * 16 + m) * 1024 + lm * 64;
    v16h aLo = build_afrag(src, hi, 0);
    v16h aHi = build_afrag(src, hi, 32);
    const _Float16* Wl = WH + 45056 + (size_t)(iter * 4 + l) * 4096;
#pragma unroll
    for (int nt = 0; nt < 4; ++nt) {
        v8f acc = {};
        v16h b0 = load_bfrag(Wl, nt * 16 + m, 64, 0, hi);
        acc = __builtin_amdgcn_wmma_f32_16x16x32_f16(false, aLo, false, b0, (short)0, acc, false, false);
        v16h b1 = load_bfrag(Wl, nt * 16 + m, 64, 32, hi);
        acc = __builtin_amdgcn_wmma_f32_16x16x32_f16(false, aHi, false, b1, (short)0, acc, false, false);
#pragma unroll
        for (int r = 0; r < 8; ++r)
            Am[(size_t)(tile * 16 + r + (hi << 3)) * 1024 + lm * 64 + nt * 16 + m] = acc[r];
    }
}

// ---------------- elementwise h-update ----------------
__global__ void k_hupdate(float* __restrict__ h, const float* __restrict__ Am,
                          const float* __restrict__ attrs, const float* __restrict__ Wz,
                          const float* __restrict__ up, const float* __restrict__ Wsc, int iter) {
    int idx = blockIdx.x * blockDim.x + threadIdx.x;
    if (idx >= NN * 1024) return;
    int c = idx & 63, lm = (idx >> 6) & 15, n = idx >> 10;
    float a  = Am[idx];
    float a0 = Am[(size_t)n * 1024 + c];
    float w2 = 0.0f, scw = 0.0f;
    const float* at = attrs + n * NZC;
#pragma unroll
    for (int z = 0; z < NZC; ++z) {
        float az = at[z];
        w2  += az * Wz[((iter * NZC + z) * 16 + lm) * 64 + c];
        scw += az * Wsc[(iter * NZC + z) * 64 + c];
    }
    float u = up[(iter * 16 + lm) * 64 + c];
    h[idx] = u * a + w2 * a0 * a + h[idx] * scw;
}

// ---------------- readout ----------------
__global__ void k_readout(const float* __restrict__ h, const float* __restrict__ Wr1,
                          const float* __restrict__ Wh2, const float* __restrict__ Wout2,
                          const int* __restrict__ batch, float* __restrict__ out, int iter) {
    int n = blockIdx.x * blockDim.x + threadIdx.x;
    if (n >= NN) return;
    float scal[64];
#pragma unroll
    for (int c = 0; c < 64; ++c) scal[c] = h[(size_t)n * 1024 + c];
    float o[30];
    if (iter == 0) {
        for (int j = 0; j < 30; ++j) {
            float s = 0.0f;
#pragma unroll
            for (int c = 0; c < 64; ++c) s += scal[c] * Wr1[c * 30 + j];
            o[j] = s;
        }
    } else {
        float hid[32];
        for (int j = 0; j < 32; ++j) {
            float s = 0.0f;
#pragma unroll
            for (int c = 0; c < 64; ++c) s += scal[c] * Wh2[c * 32 + j];
            hid[j] = siluf(s);
        }
        for (int j = 0; j < 30; ++j) {
            float s = 0.0f;
#pragma unroll
            for (int k = 0; k < 32; ++k) s += hid[k] * Wout2[k * 30 + j];
            o[j] = s;
        }
    }
    int b = batch[n];
#pragma unroll
    for (int j = 0; j < 3; ++j) {
        out[OUT_NODEE + n * 3 + j] += o[j];
        atomicAdd(&out[b * 3 + j], o[j]);
    }
#pragma unroll
    for (int k = 0; k < 9; ++k)  out[OUT_NACS + n * 9 + k]  += o[3 + k];
#pragma unroll
    for (int k = 0; k < 18; ++k) out[OUT_DIP  + n * 18 + k] += o[12 + k];
}

// ---------------- launcher ----------------
extern "C" void kernel_launch(void* const* d_in, const int* in_sizes, int n_in,
                              void* d_out, int out_size, void* d_ws, size_t ws_size,
                              hipStream_t stream) {
    const float* positions  = (const float*)d_in[0];
    const float* node_attrs = (const float*)d_in[1];
    const float* shifts     = (const float*)d_in[2];
    const float* W_embed    = (const float*)d_in[3];
    const float* atomic_e   = (const float*)d_in[4];
    const float* W_r1       = (const float*)d_in[5];
    const float* W_r2       = (const float*)d_in[6];
    const float* W_r3       = (const float*)d_in[7];
    const float* W_mix      = (const float*)d_in[8];
    const float* Wz_prod    = (const float*)d_in[9];
    const float* u_prod     = (const float*)d_in[10];
    const float* Wsc        = (const float*)d_in[11];
    const float* W_read1    = (const float*)d_in[12];
    const float* Wh2        = (const float*)d_in[13];
    const float* Wout2      = (const float*)d_in[14];
    const int*   edge_index = (const int*)d_in[15];
    const int*   batch      = (const int*)d_in[16];
    float* out = (float*)d_out;

    char* ws = (char*)d_ws;
    float*     hbuf = (float*)(ws + HOFF);
    float*     Abuf = (float*)(ws + AOFF);
    float*     Ambf = (float*)(ws + MOFF);
    float*     Ybuf = (float*)(ws + YOFF);
    float*     Fbuf = (float*)(ws + FOFF);
    _Float16*  whbf = (_Float16*)(ws + WOFF);

    // setup
    k_zero<<<(OUT_TOT + 255) / 256, 256, 0, stream>>>(out, OUT_TOT);
    k_convert_weights<<<(77824 + 255) / 256, 256, 0, stream>>>(W_r1, W_r2, W_r3, W_mix, whbf);
    k_init_h<<<(NN * 1024 + 255) / 256, 256, 0, stream>>>(hbuf, node_attrs, W_embed);
    k_init_nodes<<<(NN + 255) / 256, 256, 0, stream>>>(node_attrs, atomic_e, batch, out);
    k_edge_geom<<<(EE + 255) / 256, 256, 0, stream>>>(positions, shifts, edge_index, Ybuf, Fbuf);

    for (int iter = 0; iter < 2; ++iter) {
        k_zero<<<8192, 256, 0, stream>>>(Abuf, NN * 1024);
        k_edge_mlp<<<500, 64, 0, stream>>>(Fbuf, Ybuf, hbuf, edge_index, Abuf, whbf, iter);
        k_mix<<<(NTILE_N * 16 + 3) / 4, 128, 0, stream>>>(Abuf, Ambf, whbf, iter);
        k_hupdate<<<(NN * 1024 + 255) / 256, 256, 0, stream>>>(hbuf, Ambf, node_attrs, Wz_prod,
                                                               u_prod, Wsc, iter);
        k_readout<<<(NN + 255) / 256, 256, 0, stream>>>(hbuf, W_read1, Wh2, Wout2, batch, out, iter);
    }
}
